// CrossAttention_55576876810392
// MI455X (gfx1250) — compile-verified, matrix operand order checked by reading, not run on hardware
//
#include <hip/hip_runtime.h>

// ---------------------------------------------------------------------------
// CrossAttention block for MI455X (gfx1250, wave32, WMMA bf16 16x16x32).
//   x[8,512,32,32] -> GN(8 groups) -> QKV(512x512) -> MHA(8 heads, d=64)
//   -> proj(512x512) + residual.
// bf16 inputs to tensor cores, fp32 accumulate everywhere.
// ---------------------------------------------------------------------------

typedef __bf16 bf16;
typedef __attribute__((ext_vector_type(16))) __bf16 v16bf;
typedef __attribute__((ext_vector_type(8)))  float  v8f;

#define B_   8
#define C_   512
#define S_   1024
#define NH_  8
#define DH_  64
#define GR_  8

union Frag16 { v16bf v; uint4 q[2]; };

// Load a 16x32 bf16 operand fragment (A layout, ISA 7.12.2) from a row-major
// tile with leading dimension ld (elements). lane L: row = L&15,
// kbase = (L>>4)*8 ; VGPR0-3 hold K=kbase..kbase+7, VGPR4-7 hold K=kbase+16..+23.
// Works for B operands too: for D = A*B with B = W^T, the B operand register
// layout equals the A-layout of W's row-major storage.
__device__ inline v16bf load_frag_b16(const bf16* __restrict__ tile, int ld) {
    const int lane = threadIdx.x & 31;
    const int row  = lane & 15;
    const int kb   = (lane >> 4) << 3;
    const bf16* p = tile + row * ld + kb;
    Frag16 f;
    f.q[0] = *reinterpret_cast<const uint4*>(p);
    f.q[1] = *reinterpret_cast<const uint4*>(p + 16);
    return f.v;
}

__device__ inline v8f wmma_bf16(v16bf a, v16bf b, v8f c) {
    return __builtin_amdgcn_wmma_f32_16x16x32_bf16(
        /*neg_a=*/false, a, /*neg_b=*/false, b,
        /*c_mod=*/(short)0, c, /*reuse_a=*/false, /*reuse_b=*/false);
}

// ---------------------------------------------------------------------------
// Kernel 1: GroupNorm over (C/G=64, H*W=1024) per (b,g); writes xn as bf16
// in [B, S, C] (transposed, GEMM-friendly) layout.
// ---------------------------------------------------------------------------
__global__ void gn_kernel(const float* __restrict__ x,
                          const float* __restrict__ gw,
                          const float* __restrict__ gb,
                          bf16* __restrict__ xn) {
    const int b = blockIdx.x;
    const int g = blockIdx.y;
    const int tid = threadIdx.x;
    const float* xp = x + ((size_t)b * C_ + g * 64) * S_;   // 65536 floats

    float s = 0.f, s2 = 0.f;
    for (int i = tid; i < 64 * S_; i += 256) {
        float v = xp[i];
        s += v; s2 += v * v;
    }
    __shared__ float red0[256];
    __shared__ float red1[256];
    red0[tid] = s; red1[tid] = s2;
    __syncthreads();
    for (int off = 128; off > 0; off >>= 1) {
        if (tid < off) { red0[tid] += red0[tid + off]; red1[tid] += red1[tid + off]; }
        __syncthreads();
    }
    const float mean = red0[0] * (1.0f / 65536.0f);
    const float var  = red1[0] * (1.0f / 65536.0f) - mean * mean;
    const float inv  = rsqrtf(var + 1e-5f);

    for (int i = tid; i < 64 * S_; i += 256) {
        int c  = i >> 10;          // local channel 0..63
        int sp = i & (S_ - 1);     // spatial 0..1023
        int ch = g * 64 + c;
        float v = (xp[i] - mean) * inv * gw[ch] + gb[ch];
        xn[((size_t)b * S_ + sp) * C_ + ch] = (bf16)v;
    }
}

// ---------------------------------------------------------------------------
// Kernel 2: fp32 -> bf16 weight conversion (wq,wk,wv,wp, 512x512 each).
// ---------------------------------------------------------------------------
__global__ void cvtw_kernel(const float* __restrict__ wq,
                            const float* __restrict__ wk,
                            const float* __restrict__ wv,
                            const float* __restrict__ wp,
                            bf16* __restrict__ dst) {
    const int N = C_ * C_;   // 262144
    for (int idx = blockIdx.x * 256 + threadIdx.x; idx < 4 * N;
         idx += gridDim.x * 256) {
        int m = idx / N, r = idx - m * N;
        const float* src = (m == 0) ? wq : (m == 1) ? wk : (m == 2) ? wv : wp;
        dst[idx] = (bf16)src[r];
    }
}

// ---------------------------------------------------------------------------
// Kernel 3: QKV GEMM. M=8192, N=512, K=512. grid.z selects q/k/v.
// Block = 8 waves, wave-tile 32x32 (2x2 WMMA tiles), block-tile 128x64.
// q,k stored [B,S,C] bf16; v stored transposed [B,NH,DH,S] bf16.
// ---------------------------------------------------------------------------
__global__ void qkv_kernel(const bf16* __restrict__ xn,
                           const bf16* __restrict__ wbf,   // wq|wk|wv bf16
                           const float* __restrict__ bq,
                           const float* __restrict__ bk,
                           const float* __restrict__ bv,
                           bf16* __restrict__ qo,
                           bf16* __restrict__ ko,
                           bf16* __restrict__ vT) {
    const int which = blockIdx.z;
    const bf16* W = wbf + (size_t)which * C_ * C_;
    const float* bias = (which == 0) ? bq : (which == 1) ? bk : bv;

    const int wid  = threadIdx.x >> 5;
    const int lane = threadIdx.x & 31;
    const int m0 = blockIdx.x * 128 + (wid & 3) * 32;
    const int n0 = blockIdx.y * 64  + (wid >> 2) * 32;

    v8f c00 = {}, c01 = {}, c10 = {}, c11 = {};
    for (int k0 = 0; k0 < C_; k0 += 32) {
        v16bf a0 = load_frag_b16(xn + (size_t)m0 * C_ + k0, C_);
        v16bf a1 = load_frag_b16(xn + (size_t)(m0 + 16) * C_ + k0, C_);
        v16bf b0 = load_frag_b16(W + (size_t)n0 * C_ + k0, C_);
        v16bf b1 = load_frag_b16(W + (size_t)(n0 + 16) * C_ + k0, C_);
        c00 = wmma_bf16(a0, b0, c00);
        c01 = wmma_bf16(a0, b1, c01);
        c10 = wmma_bf16(a1, b0, c10);
        c11 = wmma_bf16(a1, b1, c11);
    }

    const int col = lane & 15;
    const int rb  = (lane >> 4) << 3;
    v8f acc[2][2] = {{c00, c01}, {c10, c11}};
    for (int i = 0; i < 2; i++)
        for (int j = 0; j < 2; j++) {
            int n = n0 + j * 16 + col;
            float bs = bias[n];
            for (int r = 0; r < 8; r++) {
                int m = m0 + i * 16 + rb + r;
                float val = acc[i][j][r] + bs;
                bf16 hv = (bf16)val;
                if (which == 2) {
                    int bb = m >> 10, sp = m & (S_ - 1);
                    int hh = n >> 6,  dd = n & 63;
                    vT[(((size_t)bb * NH_ + hh) * DH_ + dd) * S_ + sp] = hv;
                } else {
                    bf16* o = (which == 0) ? qo : ko;
                    o[(size_t)m * C_ + n] = hv;
                }
            }
        }
}

// ---------------------------------------------------------------------------
// Kernel 4: attention per (b, head, 64-row q-tile).
// Phase 1: S = scale * Q K^T  (WMMA) -> 128KB dynamic LDS (bf16 [64][1024]).
// Phase 2: row softmax (4 lanes per row, shfl_xor reductions).
// Phase 3: O = P V  (WMMA, A from LDS, B from transposed V).
// ---------------------------------------------------------------------------
__global__ void attn_kernel(const bf16* __restrict__ q,
                            const bf16* __restrict__ k,
                            const bf16* __restrict__ vT,
                            bf16* __restrict__ o) {
    extern __shared__ bf16 Psm[];           // [64][1024] = 128 KB
    const int qt = blockIdx.x;              // q-tile 0..15
    const int h  = blockIdx.y;
    const int b  = blockIdx.z;
    const int wid  = threadIdx.x >> 5;
    const int lane = threadIdx.x & 31;
    const float scale = 0.125f;             // 1/sqrt(64)
    const int q0 = qt * 64;

    const bf16* qbase = q + ((size_t)b * S_ + q0) * C_ + h * DH_;
    const bf16* kbase = k + (size_t)b * S_ * C_ + h * DH_;

    // --- phase 1: scores -> LDS ---
    v16bf afr[4][2];
    for (int mt = 0; mt < 4; mt++)
        for (int ks = 0; ks < 2; ks++)
            afr[mt][ks] = load_frag_b16(qbase + (size_t)(mt * 16) * C_ + ks * 32, C_);

    for (int nt = 0; nt < 8; nt++) {
        int n0 = wid * 128 + nt * 16;
        v16bf b0 = load_frag_b16(kbase + (size_t)n0 * C_,      C_);
        v16bf b1 = load_frag_b16(kbase + (size_t)n0 * C_ + 32, C_);
        for (int mt = 0; mt < 4; mt++) {
            v8f cc = {};
            cc = wmma_bf16(afr[mt][0], b0, cc);
            cc = wmma_bf16(afr[mt][1], b1, cc);
            int col = n0 + (lane & 15);
            int rb  = (lane >> 4) << 3;
            for (int r = 0; r < 8; r++)
                Psm[(mt * 16 + rb + r) * S_ + col] = (bf16)(cc[r] * scale);
        }
    }
    __syncthreads();

    // --- phase 2: softmax, 4 lanes per row ---
    {
        const int row = threadIdx.x >> 2;
        const int sub = threadIdx.x & 3;
        float mx = -3.0e38f;
        for (int c = sub; c < S_; c += 4)
            mx = fmaxf(mx, (float)Psm[row * S_ + c]);
        mx = fmaxf(mx, __shfl_xor(mx, 1));
        mx = fmaxf(mx, __shfl_xor(mx, 2));
        float sum = 0.f;
        for (int c = sub; c < S_; c += 4) {
            float e = __expf((float)Psm[row * S_ + c] - mx);
            sum += e;
            Psm[row * S_ + c] = (bf16)e;
        }
        sum += __shfl_xor(sum, 1);
        sum += __shfl_xor(sum, 2);
        float inv = 1.0f / sum;
        for (int c = sub; c < S_; c += 4)
            Psm[row * S_ + c] = (bf16)((float)Psm[row * S_ + c] * inv);
    }
    __syncthreads();

    // --- phase 3: O = P @ V ---
    const int mt  = wid & 3;                 // 16-row m-tile
    const int nt0 = (wid >> 2) * 2;          // two 16-col n-tiles
    const bf16* vb = vT + ((size_t)b * NH_ + h) * DH_ * S_;
    v8f o0 = {}, o1 = {};
    for (int ks = 0; ks < 32; ks++) {
        v16bf a  = load_frag_b16(Psm + (size_t)(mt * 16) * S_ + ks * 32, S_);
        v16bf b0 = load_frag_b16(vb + (size_t)(nt0 * 16) * S_ + ks * 32, S_);
        v16bf b1 = load_frag_b16(vb + (size_t)(nt0 * 16 + 16) * S_ + ks * 32, S_);
        o0 = wmma_bf16(a, b0, o0);
        o1 = wmma_bf16(a, b1, o1);
    }
    const int col = lane & 15;
    const int rb  = (lane >> 4) << 3;
    bf16* obase = o + ((size_t)b * S_ + q0) * C_ + h * DH_;
    for (int r = 0; r < 8; r++) {
        int row = mt * 16 + rb + r;
        obase[(size_t)row * C_ + nt0 * 16 + col]        = (bf16)o0[r];
        obase[(size_t)row * C_ + (nt0 + 1) * 16 + col]  = (bf16)o1[r];
    }
}

// ---------------------------------------------------------------------------
// Kernel 5: output projection + bias + residual, store fp32 [B,C,H,W].
// ---------------------------------------------------------------------------
__global__ void proj_kernel(const bf16* __restrict__ attn,
                            const bf16* __restrict__ wp,
                            const float* __restrict__ bp,
                            const float* __restrict__ x,
                            float* __restrict__ out) {
    const int wid  = threadIdx.x >> 5;
    const int lane = threadIdx.x & 31;
    const int m0 = blockIdx.x * 128 + (wid & 3) * 32;
    const int n0 = blockIdx.y * 64  + (wid >> 2) * 32;

    v8f c00 = {}, c01 = {}, c10 = {}, c11 = {};
    for (int k0 = 0; k0 < C_; k0 += 32) {
        v16bf a0 = load_frag_b16(attn + (size_t)m0 * C_ + k0, C_);
        v16bf a1 = load_frag_b16(attn + (size_t)(m0 + 16) * C_ + k0, C_);
        v16bf b0 = load_frag_b16(wp + (size_t)n0 * C_ + k0, C_);
        v16bf b1 = load_frag_b16(wp + (size_t)(n0 + 16) * C_ + k0, C_);
        c00 = wmma_bf16(a0, b0, c00);
        c01 = wmma_bf16(a0, b1, c01);
        c10 = wmma_bf16(a1, b0, c10);
        c11 = wmma_bf16(a1, b1, c11);
    }

    const int col = lane & 15;
    const int rb  = (lane >> 4) << 3;
    v8f acc[2][2] = {{c00, c01}, {c10, c11}};
    for (int i = 0; i < 2; i++)
        for (int j = 0; j < 2; j++) {
            int n = n0 + j * 16 + col;          // channel
            float bs = bp[n];
            for (int r = 0; r < 8; r++) {
                int m = m0 + i * 16 + rb + r;   // (b, s)
                int bb = m >> 10, sp = m & (S_ - 1);
                size_t oi = ((size_t)bb * C_ + n) * S_ + sp;
                out[oi] = acc[i][j][r] + bs + x[oi];
            }
        }
}

// ---------------------------------------------------------------------------
extern "C" void kernel_launch(void* const* d_in, const int* in_sizes, int n_in,
                              void* d_out, int out_size, void* d_ws, size_t ws_size,
                              hipStream_t stream) {
    const float* x   = (const float*)d_in[0];
    const float* gnw = (const float*)d_in[1];
    const float* gnb = (const float*)d_in[2];
    const float* wq  = (const float*)d_in[3];
    const float* bq  = (const float*)d_in[4];
    const float* wk  = (const float*)d_in[5];
    const float* bk  = (const float*)d_in[6];
    const float* wv  = (const float*)d_in[7];
    const float* bv  = (const float*)d_in[8];
    const float* wp  = (const float*)d_in[9];
    const float* bp  = (const float*)d_in[10];
    float* out = (float*)d_out;

    // Workspace layout (bytes); total = 2 MB weights + 5 * 8 MB = 44,040,192.
    char* ws = (char*)d_ws;
    const size_t WMAT = (size_t)C_ * C_ * sizeof(bf16);     // 512 KB each
    const size_t TEN  = (size_t)B_ * S_ * C_ * sizeof(bf16); // 8 MB each
    bf16* w_bf  = (bf16*)(ws);                    // wq|wk|wv|wp
    bf16* xn    = (bf16*)(ws + 4 * WMAT);
    bf16* qb    = (bf16*)(ws + 4 * WMAT + 1 * TEN);
    bf16* kb    = (bf16*)(ws + 4 * WMAT + 2 * TEN);
    bf16* vT    = (bf16*)(ws + 4 * WMAT + 3 * TEN);
    bf16* ao    = (bf16*)(ws + 4 * WMAT + 4 * TEN);

    gn_kernel<<<dim3(B_, GR_), 256, 0, stream>>>(x, gnw, gnb, xn);
    cvtw_kernel<<<512, 256, 0, stream>>>(wq, wk, wv, wp, w_bf);
    qkv_kernel<<<dim3(B_ * S_ / 128, C_ / 64, 3), 256, 0, stream>>>(
        xn, w_bf, bq, bk, bv, qb, kb, vT);
    attn_kernel<<<dim3(S_ / 64, NH_, B_), 256, 64 * S_ * sizeof(bf16), stream>>>(
        qb, kb, vT, ao);
    proj_kernel<<<dim3(B_ * S_ / 128, C_ / 64), 256, 0, stream>>>(
        ao, w_bf + 3 * (size_t)C_ * C_, bp, x, out);
}
